// FeatureMoE_3925600108737
// MI455X (gfx1250) — matrix operand order repaired and checked by reading, not compiled, not run on hardware
//
#include <hip/hip_runtime.h>
#include <hip/hip_bf16.h>

typedef __bf16 v16bf __attribute__((ext_vector_type(16)));
typedef float  v8f   __attribute__((ext_vector_type(8)));

union BF16x16 { uint4 q[2]; v16bf v; };

#define DF 256
#define HD 256
#define NE 8
#define NF 64
#define NB 256
#define MTOT (NB * NF)          // 16384 rows
#define MT 32                   // rows per block
#define LDA 264                 // padded LDS row stride (bf16 elements)
#define BN_EPS 1e-3f

// ---------------------------------------------------------------------------
// helpers: CDNA5 async global->LDS staging (ASYNCcnt-tracked)
// ---------------------------------------------------------------------------
__device__ __forceinline__ unsigned lds_off(const void* p) {
    // generic LDS pointer: low 32 bits are the LDS byte offset
    return (unsigned)(unsigned long long)p;
}

__device__ __forceinline__ void wait_async0() {
#if __has_builtin(__builtin_amdgcn_s_wait_asynccnt)
    __builtin_amdgcn_s_wait_asynccnt(0);
#else
    asm volatile("s_wait_asynccnt 0x0" ::: "memory");
#endif
}

// copy 16KB contiguous: 256 threads x 4 async b128 issues.
// inst offset is added to BOTH the LDS and global address (ISA 08 §4.4).
__device__ __forceinline__ void async_copy_16k(const __bf16* gsrc, unsigned ldst, int tid)
{
    unsigned l = ldst + tid * 16;
    unsigned g = (unsigned)(tid * 16);
    asm volatile(
        "global_load_async_to_lds_b128 %0, %1, %2 offset:0\n\t"
        "global_load_async_to_lds_b128 %0, %1, %2 offset:4096\n\t"
        "global_load_async_to_lds_b128 %0, %1, %2 offset:8192\n\t"
        "global_load_async_to_lds_b128 %0, %1, %2 offset:12288"
        :: "v"(l), "v"(g), "s"(gsrc) : "memory");
}

// stage 32x256 bf16 x-tile into row-padded LDS (64B per thread, 4 async b128)
__device__ __forceinline__ void async_stage_xtile(const __bf16* xb, unsigned aact,
                                                  int mBase, int tid)
{
    int row = tid >> 3, seg = tid & 7;
    unsigned l = aact + (unsigned)(row * (LDA * 2) + seg * 64);
    unsigned g = (unsigned)((mBase + row) * (DF * 2) + seg * 64);
    asm volatile(
        "global_load_async_to_lds_b128 %0, %1, %2 offset:0\n\t"
        "global_load_async_to_lds_b128 %0, %1, %2 offset:16\n\t"
        "global_load_async_to_lds_b128 %0, %1, %2 offset:32\n\t"
        "global_load_async_to_lds_b128 %0, %1, %2 offset:48"
        :: "v"(l), "v"(g), "s"(xb) : "memory");
}

// ---------------------------------------------------------------------------
// Kernel 1: routing.  feat = mean_b x[b,f,:]; w = softmax(feat @ Wr + br)
// ---------------------------------------------------------------------------
__global__ __launch_bounds__(256) void routing_kernel(
    const float* __restrict__ x, const float* __restrict__ Wr,
    const float* __restrict__ br, float* __restrict__ wout)
{
    int f = blockIdx.x;
    int d = threadIdx.x;
    const float* xp = x + f * DF + d;
    float s = 0.f;
    for (int b = 0; b < NB; ++b) s += xp[(size_t)b * (NF * DF)];
    float feat = s * (1.0f / (float)NB);

    __shared__ float red[256][NE];
#pragma unroll
    for (int e = 0; e < NE; ++e) red[d][e] = feat * Wr[d * NE + e];
    __syncthreads();
    for (int off = 128; off > 0; off >>= 1) {
        if (d < off) {
#pragma unroll
            for (int e = 0; e < NE; ++e) red[d][e] += red[d + off][e];
        }
        __syncthreads();
    }
    if (d == 0) {
        float lg[NE], mx = -1e30f;
#pragma unroll
        for (int e = 0; e < NE; ++e) { lg[e] = red[0][e] + br[e]; mx = fmaxf(mx, lg[e]); }
        float sum = 0.f;
#pragma unroll
        for (int e = 0; e < NE; ++e) { lg[e] = expf(lg[e] - mx); sum += lg[e]; }
        float inv = 1.0f / sum;
#pragma unroll
        for (int e = 0; e < NE; ++e) wout[f * NE + e] = lg[e] * inv;
    }
}

// ---------------------------------------------------------------------------
// Kernel 2: fold bias + batchnorm into per-column affine (scale, shift).
// ---------------------------------------------------------------------------
__global__ void bncoef_kernel(
    const float* __restrict__ b1, const float* __restrict__ g1,
    const float* __restrict__ be1, const float* __restrict__ m1, const float* __restrict__ v1,
    const float* __restrict__ b2, const float* __restrict__ g2,
    const float* __restrict__ be2, const float* __restrict__ m2, const float* __restrict__ v2,
    const float* __restrict__ bo,
    float* __restrict__ sArr, float* __restrict__ tArr)
{
    int idx = blockIdx.x * blockDim.x + threadIdx.x;
    if (idx >= 3 * NE * HD) return;
    int layer = idx >> 11;
    int rem   = idx & 2047;
    float s, t;
    if (layer == 0) {
        float sc = g1[rem] * rsqrtf(v1[rem] + BN_EPS);
        s = sc; t = (b1[rem] - m1[rem]) * sc + be1[rem];
    } else if (layer == 1) {
        float sc = g2[rem] * rsqrtf(v2[rem] + BN_EPS);
        s = sc; t = (b2[rem] - m2[rem]) * sc + be2[rem];
    } else {
        s = 1.0f; t = bo[rem];
    }
    sArr[idx] = s; tArr[idx] = t;
}

// ---------------------------------------------------------------------------
// Kernel 3: x f32 -> bf16
// ---------------------------------------------------------------------------
__global__ void xconv_kernel(const float* __restrict__ x, __bf16* __restrict__ xb)
{
    int i = (blockIdx.x * blockDim.x + threadIdx.x) * 4;
    float4 v = *(const float4*)(x + i);
    xb[i + 0] = (__bf16)v.x;
    xb[i + 1] = (__bf16)v.y;
    xb[i + 2] = (__bf16)v.z;
    xb[i + 3] = (__bf16)v.w;
}

// ---------------------------------------------------------------------------
// Kernel 4: weights f32 -> bf16, pre-swizzled into WMMA B-fragment order.
// wsw[layer][e][kc(8)][nt(16)][lane(32)][m(16)]:
//   n = nt*16 + (lane&15), K = kc*32 + 8*(lane>>4) + (m&7) + 16*(m>>3)
// ---------------------------------------------------------------------------
__global__ void wswz_kernel(const float* __restrict__ W1, const float* __restrict__ W2,
                            const float* __restrict__ Wo, __bf16* __restrict__ wsw)
{
    int idx = blockIdx.x * blockDim.x + threadIdx.x;
    int m    = idx & 15;
    int lane = (idx >> 4) & 31;
    int nt   = (idx >> 9) & 15;
    int kc   = (idx >> 13) & 7;
    int e    = (idx >> 16) & 7;
    int layer= idx >> 19;
    int k = kc * 32 + ((lane >> 4) << 3) + (m & 7) + ((m >> 3) << 4);
    int n = nt * 16 + (lane & 15);
    const float* W = (layer == 0) ? W1 : ((layer == 1) ? W2 : Wo);
    wsw[idx] = (__bf16)W[(size_t)e * (HD * HD) + k * HD + n];
}

// ---------------------------------------------------------------------------
// Main fused MoE kernel.  512 blocks x 256 threads (8 waves, wave32).
// Waves tiled wm(2) x wn(4): 16x64 output each.
// B chunks double-buffered via global_load_async_to_lds_b128 (ASYNCcnt),
// overlapped with v_wmma_f32_16x16x32_bf16 on the other buffer.
// ---------------------------------------------------------------------------
__device__ __forceinline__ void run_layer(
    const __bf16* __restrict__ Asrc,        // LDS activations [MT][LDA]
    const __bf16* __restrict__ wswL,        // global, swizzled, 65536 bf16
    const __bf16* __restrict__ Bs0, const __bf16* __restrict__ Bs1,
    unsigned bsoff0, unsigned bsoff1,
    v8f acc[4], int wm, int wn, int lane, int tid)
{
    const int khalf = (lane >> 4) * 8;
    const int rowA  = wm * 16 + (lane & 15);

    // prologue: chunk 0 -> buf0 (also covers any pending x-tile staging)
    async_copy_16k(wswL, bsoff0, tid);
    wait_async0();
    __syncthreads();

#pragma unroll
    for (int kc = 0; kc < 8; ++kc) {
        if (kc < 7)   // prefetch next chunk into the other buffer
            async_copy_16k(wswL + (kc + 1) * 8192, (kc & 1) ? bsoff0 : bsoff1, tid);

        const __bf16* bsrc = (kc & 1) ? Bs1 : Bs0;
        // A fragment: elems 0-7 = K[kh..kh+7], elems 8-15 = K[16+kh..16+kh+7]
        BF16x16 a;
        const __bf16* ap = Asrc + rowA * LDA + kc * 32 + khalf;
        a.q[0] = *(const uint4*)ap;
        a.q[1] = *(const uint4*)(ap + 16);
#pragma unroll
        for (int t = 0; t < 4; ++t) {
            BF16x16 b;
            const __bf16* bp = bsrc + (((wn * 4 + t) * 32 + lane) << 4);
            b.q[0] = ((const uint4*)bp)[0];
            b.q[1] = ((const uint4*)bp)[1];
            acc[t] = __builtin_amdgcn_wmma_f32_16x16x32_bf16(
                false, a.v, false, b.v, (short)0, acc[t], false, false);
        }
        wait_async0();
        __syncthreads();   // prefetched buffer ready; all reads of cur done
    }
}

__global__ __launch_bounds__(256) void moe_main_kernel(
    const __bf16* __restrict__ xb, const __bf16* __restrict__ wsw,
    const float* __restrict__ sArr, const float* __restrict__ tArr,
    const float* __restrict__ wroute, float* __restrict__ out)
{
    __shared__ __align__(16) __bf16 Aact[MT * LDA];     // 16896 B, in-place acts
    __shared__ __align__(16) __bf16 Bs[2][8192];        // 2 x 16384 B ping-pong

    const int tid  = threadIdx.x;
    const int lane = tid & 31;
    const int wave = tid >> 5;
    const int wm   = wave >> 2;     // 0..1
    const int wn   = wave & 3;      // 0..3
    const int mBase = blockIdx.x * MT;

    const unsigned aactOff = lds_off(&Aact[0]);
    const unsigned bsoff0  = lds_off(&Bs[0][0]);
    const unsigned bsoff1  = lds_off(&Bs[1][0]);

    v8f oacc[4] = {};
    const int colBase = wn * 64 + (lane & 15);
    const int rowLoc  = wm * 16 + ((lane >> 4) << 3);   // + r

    for (int e = 0; e < NE; ++e) {
        // restage x tile (async; completion folded into layer-0 prologue wait)
        async_stage_xtile(xb, aactOff, mBase, tid);

        // ---------------- layer 0: x @ W1 -> bn/relu -> Aact (in place) -----
        {
            v8f acc[4] = {};
            run_layer(Aact, wsw + (size_t)(0 * NE + e) * (HD * HD),
                      Bs[0], Bs[1], bsoff0, bsoff1, acc, wm, wn, lane, tid);
#pragma unroll
            for (int t = 0; t < 4; ++t) {
                int col = colBase + t * 16;
                float s  = sArr[(0 * NE + e) * HD + col];
                float tt = tArr[(0 * NE + e) * HD + col];
#pragma unroll
                for (int r = 0; r < 8; ++r) {
                    float v = acc[t][r] * s + tt;
                    v = v > 0.f ? v : 0.f;
                    Aact[(rowLoc + r) * LDA + col] = (__bf16)v;
                }
            }
        }
        // ---------------- layer 1: Aact @ W2 -> bn/relu -> Aact -------------
        {
            v8f acc[4] = {};
            run_layer(Aact, wsw + (size_t)(1 * NE + e) * (HD * HD),
                      Bs[0], Bs[1], bsoff0, bsoff1, acc, wm, wn, lane, tid);
#pragma unroll
            for (int t = 0; t < 4; ++t) {
                int col = colBase + t * 16;
                float s  = sArr[(1 * NE + e) * HD + col];
                float tt = tArr[(1 * NE + e) * HD + col];
#pragma unroll
                for (int r = 0; r < 8; ++r) {
                    float v = acc[t][r] * s + tt;
                    v = v > 0.f ? v : 0.f;
                    Aact[(rowLoc + r) * LDA + col] = (__bf16)v;
                }
            }
        }
        // ---------------- layer 2: Aact @ Wo + bo, routed accumulate --------
        {
            v8f acc[4] = {};
            run_layer(Aact, wsw + (size_t)(2 * NE + e) * (HD * HD),
                      Bs[0], Bs[1], bsoff0, bsoff1, acc, wm, wn, lane, tid);
            float wv[8];
#pragma unroll
            for (int r = 0; r < 8; ++r) {
                int f = (mBase + rowLoc + r) & (NF - 1);
                wv[r] = wroute[f * NE + e];
            }
#pragma unroll
            for (int t = 0; t < 4; ++t) {
                int col = colBase + t * 16;
                float tt = tArr[(2 * NE + e) * HD + col];
#pragma unroll
                for (int r = 0; r < 8; ++r)
                    oacc[t][r] += wv[r] * (acc[t][r] + tt);
            }
        }
    }

    // final store
#pragma unroll
    for (int t = 0; t < 4; ++t) {
        int col = colBase + t * 16;
#pragma unroll
        for (int r = 0; r < 8; ++r)
            out[(size_t)(mBase + rowLoc + r) * DF + col] = oacc[t][r];
    }
}

// ---------------------------------------------------------------------------
// launch
// ---------------------------------------------------------------------------
extern "C" void kernel_launch(void* const* d_in, const int* in_sizes, int n_in,
                              void* d_out, int out_size, void* d_ws, size_t ws_size,
                              hipStream_t stream)
{
    (void)in_sizes; (void)n_in; (void)out_size; (void)ws_size;
    const float* x   = (const float*)d_in[0];
    const float* Wr  = (const float*)d_in[1];
    const float* br  = (const float*)d_in[2];
    const float* W1  = (const float*)d_in[3];
    const float* b1  = (const float*)d_in[4];
    const float* g1  = (const float*)d_in[5];
    const float* be1 = (const float*)d_in[6];
    const float* m1  = (const float*)d_in[7];
    const float* v1  = (const float*)d_in[8];
    const float* W2  = (const float*)d_in[9];
    const float* b2  = (const float*)d_in[10];
    const float* g2  = (const float*)d_in[11];
    const float* be2 = (const float*)d_in[12];
    const float* m2  = (const float*)d_in[13];
    const float* v2  = (const float*)d_in[14];
    const float* Wo  = (const float*)d_in[15];
    const float* bo  = (const float*)d_in[16];
    float* out = (float*)d_out;

    // workspace layout (bytes):
    //   [0, 2048)            routing weights (64*8 f32)
    //   [2048, 26624)        sArr (3*8*256 f32)
    //   [26624, 51200)       tArr
    //   [51200, +8388608)    xb  (bf16 x)
    //   [+.., +3145728)      wsw (bf16 swizzled weights)
    char* ws = (char*)d_ws;
    float*  wroute = (float*)(ws);
    float*  sArr   = (float*)(ws + 2048);
    float*  tArr   = (float*)(ws + 26624);
    __bf16* xb     = (__bf16*)(ws + 51200);
    __bf16* wsw    = (__bf16*)(ws + 51200 + 8388608);

    routing_kernel<<<NF, 256, 0, stream>>>(x, Wr, br, wroute);
    bncoef_kernel<<<24, 256, 0, stream>>>(b1, g1, be1, m1, v1,
                                          b2, g2, be2, m2, v2, bo, sArr, tArr);
    xconv_kernel<<<(NB * NF * DF) / (4 * 256), 256, 0, stream>>>(x, xb);
    wswz_kernel<<<(3 * NE * HD * HD) / 256, 256, 0, stream>>>(W1, W2, Wo, wsw);
    moe_main_kernel<<<MTOT / MT, 256, 0, stream>>>(xb, wsw, sArr, tArr, wroute, out);
}
